// CausalSelfAttention_53403623358534
// MI455X (gfx1250) — matrix operand order repaired and checked by reading, not compiled, not run on hardware
//
#include <hip/hip_runtime.h>

// Problem constants (from reference)
#define TT   2048
#define CCH  2048
#define HH   16
#define DD   128
#define N3   6144
#define BTC  (4 * 2048 * 2048)   // B*T*C elements

typedef __attribute__((ext_vector_type(16))) __bf16 v16bf;
typedef __attribute__((ext_vector_type(8)))  float  v8f;
typedef __attribute__((ext_vector_type(2)))  __bf16 v2bf;

union V16 { v16bf v; unsigned u[8]; uint4 q[2]; };

// ---- fp32 -> packed bf16 conversion -------------------------------------
// Prefer the hardware packed convert (v_cvt_pk_bf16_f32) when the builtin
// exists; fall back to a software round-to-nearest-even bit trick.
#if defined(__has_builtin)
#if __has_builtin(__builtin_amdgcn_cvt_pk_bf16_f32)
#define HAVE_CVT_PK_BF16 1
#endif
#endif

__device__ __forceinline__ unsigned pk2(float a, float b) {
#ifdef HAVE_CVT_PK_BF16
    v2bf p = __builtin_amdgcn_cvt_pk_bf16_f32(a, b);
    return __builtin_bit_cast(unsigned, p);
#else
    unsigned ua = __float_as_uint(a);
    unsigned ub = __float_as_uint(b);
    unsigned ra = (ua + 0x7FFFu + ((ua >> 16) & 1u)) >> 16;
    unsigned rb = (ub + 0x7FFFu + ((ub >> 16) & 1u)) & 0xFFFF0000u;
    return ra | rb;
#endif
}

#define ASTR 40   // LDS row stride in bf16 elems for 32-wide K tiles (80B: 16B-aligned, conflict-free)

// ---------------------------------------------------------------------------
// Shared GEMM core: C[128x128] = X[row0.., 2048] * W[.., col0..], bf16 WMMA,
// fp32 accumulate, double-buffered LDS. 8 waves; wave -> 2 row-tiles x 4 col
// tiles at {c0, c0+16, c0+64, c0+80} (c0 = 32*(wave>>2)) so RoPE pairs stay
// in-lane for the QKV epilogue.
// ---------------------------------------------------------------------------
__device__ __forceinline__ void gemm_core(
    const float* __restrict__ X, const float* __restrict__ W, int N,
    int row0, int col0,
    unsigned short* __restrict__ As, unsigned short* __restrict__ Bs,
    v8f (&acc)[2][4])
{
    const int tid  = threadIdx.x;
    const int lane = tid & 31;
    const int wave = tid >> 5;
    const int l15  = lane & 15;
    const int hi   = lane >> 4;
    const int wr   = wave & 3;
    const int c0   = (wave >> 2) * 32;

    #pragma unroll
    for (int i = 0; i < 2; i++)
        #pragma unroll
        for (int j = 0; j < 4; j++)
            #pragma unroll
            for (int e = 0; e < 8; e++) acc[i][j][e] = 0.0f;

    // staging thread mapping
    const int am = tid >> 1;            // A: row in tile (0..127)
    const int ak = (tid & 1) * 16;      // A: k offset (0 or 16)
    const int bn = tid & 127;           // B: col in tile (0..127)
    const int bk = (tid >> 7) * 16;     // B: k offset (0 or 16)

    const float* xp = X + (size_t)(row0 + am) * (size_t)CCH + ak;
    const float* wp = W + (size_t)bk * (size_t)N + (size_t)(col0 + bn);

    auto loadA = [&](int kb, unsigned short* Ab) {
        const float* p = xp + kb;
        float4 f0 = *(const float4*)(p + 0);
        float4 f1 = *(const float4*)(p + 4);
        float4 f2 = *(const float4*)(p + 8);
        float4 f3 = *(const float4*)(p + 12);
        uint4* d4 = (uint4*)(Ab + am * ASTR + ak);
        d4[0] = make_uint4(pk2(f0.x, f0.y), pk2(f0.z, f0.w), pk2(f1.x, f1.y), pk2(f1.z, f1.w));
        d4[1] = make_uint4(pk2(f2.x, f2.y), pk2(f2.z, f2.w), pk2(f3.x, f3.y), pk2(f3.z, f3.w));
    };
    auto loadB = [&](int kb, unsigned short* Bb) {   // stores transposed: Bb[n][k]
        const float* p = wp + (size_t)kb * (size_t)N;
        unsigned u[8];
        #pragma unroll
        for (int j = 0; j < 8; j++) {
            float fa = p[(size_t)(2 * j)     * (size_t)N];
            float fb = p[(size_t)(2 * j + 1) * (size_t)N];
            u[j] = pk2(fa, fb);
        }
        uint4* d4 = (uint4*)(Bb + bn * ASTR + bk);
        d4[0] = make_uint4(u[0], u[1], u[2], u[3]);
        d4[1] = make_uint4(u[4], u[5], u[6], u[7]);
    };

    loadA(0, As);
    loadB(0, Bs);
    __syncthreads();

    for (int kb = 0; kb < CCH; kb += 32) {
        const int buf = (kb >> 5) & 1;
        unsigned short* Ab = As + buf * 128 * ASTR;
        unsigned short* Bb = Bs + buf * 128 * ASTR;
        if (kb + 32 < CCH) {                    // prefetch next tile into other buffer
            loadA(kb + 32, As + (buf ^ 1) * 128 * ASTR);
            loadB(kb + 32, Bs + (buf ^ 1) * 128 * ASTR);
        }
        v16bf af[2], bfr[4];
        #pragma unroll
        for (int mi = 0; mi < 2; mi++) {
            // A 16x32 bf16 fragment: elems 0..7 <-> k=8hi.., elems 8..15 <-> k=8hi+16..
            const unsigned short* ap = Ab + (32 * wr + 16 * mi + l15) * ASTR + 8 * hi;
            V16 u; u.q[0] = *(const uint4*)ap; u.q[1] = *(const uint4*)(ap + 16);
            af[mi] = u.v;
        }
        #pragma unroll
        for (int ci = 0; ci < 4; ci++) {
            // B 32x16 bf16 fragment: lane col = l15, elems 0..7 <-> k=8hi.., elems 8..15 <-> k=8hi+16..
            const int ct = c0 + ((ci & 1) * 16) + ((ci >> 1) * 64);
            const unsigned short* bp = Bb + (ct + l15) * ASTR + 8 * hi;
            V16 u; u.q[0] = *(const uint4*)bp; u.q[1] = *(const uint4*)(bp + 16);
            bfr[ci] = u.v;
        }
        #pragma unroll
        for (int mi = 0; mi < 2; mi++)
            #pragma unroll
            for (int ci = 0; ci < 4; ci++)
                acc[mi][ci] = __builtin_amdgcn_wmma_f32_16x16x32_bf16(
                    false, af[mi], false, bfr[ci], (short)0, acc[mi][ci], false, false);
        __syncthreads();
    }
}

// ---------------------------------------------------------------------------
// Kernel 1: QKV = x@W_attn + b, fused RoPE, scatter to q(ws) / k,v (outputs)
// in [B,H,T,D] layout. One 128-col block == one head of q, k or v.
// ---------------------------------------------------------------------------
__launch_bounds__(256)
__global__ void qkv_rope_kernel(const float* __restrict__ X,
                                const float* __restrict__ W,
                                const float* __restrict__ bias,
                                float* __restrict__ qws,
                                float* __restrict__ kout,
                                float* __restrict__ vout)
{
    __shared__ __align__(16) unsigned short As[2 * 128 * ASTR];
    __shared__ __align__(16) unsigned short Bs[2 * 128 * ASTR];
    const int nb = blockIdx.x;   // 0..47  (16 q heads, 16 k heads, 16 v heads)
    const int mb = blockIdx.y;   // 0..63

    v8f acc[2][4];
    gemm_core(X, W, N3, mb * 128, nb * 128, As, Bs, acc);

    const int tid = threadIdx.x, lane = tid & 31, wave = tid >> 5;
    const int l15 = lane & 15, hi = lane >> 4;
    const int wr = wave & 3, c0 = (wave >> 2) * 32;

    // bias (column-wise)
    #pragma unroll
    for (int ci = 0; ci < 4; ci++) {
        const int ct = c0 + ((ci & 1) * 16) + ((ci >> 1) * 64);
        const float bv = bias[nb * 128 + ct + l15];
        #pragma unroll
        for (int mi = 0; mi < 2; mi++)
            #pragma unroll
            for (int e = 0; e < 8; e++) acc[mi][ci][e] += bv;
    }

    if (nb < 32) {
        // q or k head: apply rotary. Pair tiles (0,2) and (1,3): d and d+64 in-lane.
        const int head = (nb < 16) ? nb : (nb - 16);
        float* __restrict__ dst = (nb < 16) ? qws : kout;
        #pragma unroll
        for (int pi = 0; pi < 2; pi++) {
            const int dlo = c0 + 16 * pi + l15;            // in [0,64)
            const float invf = __powf(10000.0f, -(float)dlo * (1.0f / 64.0f));
            #pragma unroll
            for (int mi = 0; mi < 2; mi++) {
                #pragma unroll
                for (int e = 0; e < 8; e++) {
                    const int r    = mb * 128 + 32 * wr + 16 * mi + e + 8 * hi;
                    const int bidx = r >> 11;
                    const int t    = r & 2047;
                    const float ang = (float)t * invf;
                    const float cth = __cosf(ang), sth = __sinf(ang);
                    const float a0 = acc[mi][pi][e];
                    const float a1 = acc[mi][pi + 2][e];
                    const size_t o = ((size_t)((bidx * HH + head) * TT + t)) * DD;
                    dst[o + dlo]      = a0 * cth - a1 * sth;   // d < 64: x*cos - x[d+64]*sin
                    dst[o + dlo + 64] = a1 * cth + a0 * sth;   // d >=64: x*cos + x[d-64]*sin
                }
            }
        }
    } else {
        // v head: bias only
        const int head = nb - 32;
        #pragma unroll
        for (int ci = 0; ci < 4; ci++) {
            const int ct = c0 + ((ci & 1) * 16) + ((ci >> 1) * 64);
            const int d = ct + l15;
            #pragma unroll
            for (int mi = 0; mi < 2; mi++)
                #pragma unroll
                for (int e = 0; e < 8; e++) {
                    const int r    = mb * 128 + 32 * wr + 16 * mi + e + 8 * hi;
                    const int bidx = r >> 11, t = r & 2047;
                    vout[((size_t)((bidx * HH + head) * TT + t)) * DD + d] = acc[mi][ci][e];
                }
        }
    }
}

// ---------------------------------------------------------------------------
// Kernel 2: causal flash attention, transposed-S formulation.
// Block = (b, h, 128-query tile); wave owns 16 queries (lane col = query).
// ---------------------------------------------------------------------------
#define KSTR 136   // 128 + 8 pad (272B rows: 16B aligned, conflict-free)
#define VSTR 72    // 64 + 8 pad  (144B rows)

__launch_bounds__(256)
__global__ void attn_kernel(const float* __restrict__ Qg,
                            const float* __restrict__ Kg,
                            const float* __restrict__ Vg,
                            float* __restrict__ Yg)
{
    __shared__ __align__(16) union {
        struct { unsigned short Ks[64 * KSTR]; unsigned short Vt[128 * VSTR]; } kv;
        float ys[128 * 128];
    } sm;

    const int qb = blockIdx.x, h = blockIdx.y, b = blockIdx.z;
    const int tid = threadIdx.x, lane = tid & 31, wave = tid >> 5;
    const int l15 = lane & 15, hi = lane >> 4;
    const size_t hoff = (size_t)(b * HH + h) * TT * DD;
    const float* Kp = Kg + hoff;
    const float* Vp = Vg + hoff;
    const int q0  = qb * 128;
    const int myq = q0 + wave * 16 + l15;   // query owned by this lane (col)

    // Preload Q as B-operand fragments: qf[c] covers d = 32c + 16hi + j
    v16bf qf[4];
    {
        const float* qrow = Qg + hoff + (size_t)myq * DD;
        #pragma unroll
        for (int c = 0; c < 4; c++) {
            const float* p = qrow + 32 * c + 16 * hi;
            V16 u;
            #pragma unroll
            for (int j = 0; j < 4; j++) {
                float4 f = *(const float4*)(p + 4 * j);
                u.u[2 * j]     = pk2(f.x, f.y);
                u.u[2 * j + 1] = pk2(f.z, f.w);
            }
            qf[c] = u.v;
        }
    }

    v8f oacc[8];   // O^T: 8 d-tiles x (lane col = q)
    #pragma unroll
    for (int dt = 0; dt < 8; dt++)
        #pragma unroll
        for (int e = 0; e < 8; e++) oacc[dt][e] = 0.0f;
    float mrow = -1e30f, lrow = 0.0f;
    const float scale = 0.08838834764831845f;   // 1/sqrt(128)

    const int key  = tid >> 2;   // K staging: 0..63
    const int part = tid & 3;    // K staging: d quarter

    const int ntiles = 2 * (qb + 1);   // 64-key tiles up to the causal block bound
    for (int kt = 0; kt < ntiles; kt++) {
        const int kk0 = kt * 64;
        // --- cooperative stage: K row-major bf16 ---
        const float* kp = Kp + (size_t)(kk0 + key) * DD + 32 * part;
        {
            unsigned u[16];
            #pragma unroll
            for (int j = 0; j < 8; j++) {
                float4 f = *(const float4*)(kp + 4 * j);
                u[2 * j]     = pk2(f.x, f.y);
                u[2 * j + 1] = pk2(f.z, f.w);
            }
            uint4* d4 = (uint4*)(sm.kv.Ks + key * KSTR + 32 * part);
            d4[0] = make_uint4(u[0],  u[1],  u[2],  u[3]);
            d4[1] = make_uint4(u[4],  u[5],  u[6],  u[7]);
            d4[2] = make_uint4(u[8],  u[9],  u[10], u[11]);
            d4[3] = make_uint4(u[12], u[13], u[14], u[15]);
        }
        // --- cooperative stage: V transposed in registers, b128 LDS stores ---
        // thread owns 8 keys (8*wave..+7) x 4 d (4*lane..+3); adjacent keys are
        // adjacent shorts in the transposed row -> pack with pk2, store uint4.
        const float* vp = Vp + (size_t)(kk0 + 8 * wave) * DD + 4 * lane;
        {
            float4 f[8];
            #pragma unroll
            for (int j = 0; j < 8; j++) f[j] = *(const float4*)(vp + (size_t)j * DD);
            #pragma unroll
            for (int c = 0; c < 4; c++) {
                const int d = 4 * lane + c;
                uint4 s = make_uint4(
                    pk2(((const float*)&f[0])[c], ((const float*)&f[1])[c]),
                    pk2(((const float*)&f[2])[c], ((const float*)&f[3])[c]),
                    pk2(((const float*)&f[4])[c], ((const float*)&f[5])[c]),
                    pk2(((const float*)&f[6])[c], ((const float*)&f[7])[c]));
                *(uint4*)(sm.kv.Vt + d * VSTR + 8 * wave) = s;
            }
        }
        // prefetch next tile toward the caches while this one computes
        if (kt + 1 < ntiles) {
            __builtin_prefetch(kp + (size_t)64 * DD, 0, 3);
            __builtin_prefetch(vp + (size_t)64 * DD, 0, 3);
        }
        __syncthreads();

        // --- S^T = K * Q^T  (rows = keys, cols = queries) ---
        v8f st[4];
        #pragma unroll
        for (int m = 0; m < 4; m++) {
            #pragma unroll
            for (int e = 0; e < 8; e++) st[m][e] = 0.0f;
            #pragma unroll
            for (int c = 0; c < 4; c++) {
                const unsigned short* ap = sm.kv.Ks + (16 * m + l15) * KSTR + 32 * c + 16 * hi;
                V16 u; u.q[0] = *(const uint4*)ap; u.q[1] = *(const uint4*)(ap + 8);
                st[m] = __builtin_amdgcn_wmma_f32_16x16x32_bf16(
                    false, u.v, false, qf[c], (short)0, st[m], false, false);
            }
        }

        // --- scale + causal mask + online softmax (per-lane = per-query) ---
        float tmax = -3.0e38f;
        #pragma unroll
        for (int m = 0; m < 4; m++)
            #pragma unroll
            for (int e = 0; e < 8; e++) {
                float s = st[m][e] * scale;
                const int kidx = kk0 + 16 * m + e + 8 * hi;
                s = (kidx <= myq) ? s : -3.0e38f;
                st[m][e] = s;
                tmax = fmaxf(tmax, s);
            }
        tmax = fmaxf(tmax, __shfl_xor(tmax, 16, 32));
        const float mnew = fmaxf(mrow, tmax);
        const float corr = __expf(mrow - mnew);
        float tsum = 0.0f;
        #pragma unroll
        for (int m = 0; m < 4; m++)
            #pragma unroll
            for (int e = 0; e < 8; e++) {
                const float pe = __expf(st[m][e] - mnew);
                st[m][e] = pe;
                tsum += pe;
            }
        tsum += __shfl_xor(tsum, 16, 32);
        lrow = lrow * corr + tsum;
        mrow = mnew;
        #pragma unroll
        for (int dt = 0; dt < 8; dt++)
            #pragma unroll
            for (int e = 0; e < 8; e++) oacc[dt][e] *= corr;

        // --- P^T -> bf16 B fragments: half-wave swap via shfl_xor(16) ---
        v16bf pf[2];
        #pragma unroll
        for (int c2 = 0; c2 < 2; c2++) {
            float lo[8], hh[8];
            #pragma unroll
            for (int e = 0; e < 8; e++) {
                const float o0 = st[2 * c2][e];
                const float o1 = st[2 * c2 + 1][e];
                const float x0 = __shfl_xor(o0, 16, 32);
                const float x1 = __shfl_xor(o1, 16, 32);
                lo[e] = hi ? x1 : o0;
                hh[e] = hi ? o1 : x0;
            }
            V16 u;
            #pragma unroll
            for (int j = 0; j < 4; j++) {
                u.u[j]     = pk2(lo[2 * j], lo[2 * j + 1]);
                u.u[4 + j] = pk2(hh[2 * j], hh[2 * j + 1]);
            }
            pf[c2] = u.v;
        }

        // --- O^T += V^T * P^T ---
        #pragma unroll
        for (int dt = 0; dt < 8; dt++)
            #pragma unroll
            for (int c2 = 0; c2 < 2; c2++) {
                const unsigned short* ap = sm.kv.Vt + (16 * dt + l15) * VSTR + 32 * c2 + 16 * hi;
                V16 u; u.q[0] = *(const uint4*)ap; u.q[1] = *(const uint4*)(ap + 8);
                oacc[dt] = __builtin_amdgcn_wmma_f32_16x16x32_bf16(
                    false, u.v, false, pf[c2], (short)0, oacc[dt], false, false);
            }
        __syncthreads();
    }

    // --- normalize, transpose back through LDS, coalesced store to y ws ---
    const float inv = 1.0f / lrow;
    #pragma unroll
    for (int dt = 0; dt < 8; dt++)
        #pragma unroll
        for (int e = 0; e < 8; e++)
            sm.ys[(wave * 16 + l15) * 128 + 16 * dt + e + 8 * hi] = oacc[dt][e] * inv;
    __syncthreads();
    {
        const int row = tid >> 1;
        const int half = tid & 1;
        const float* src = sm.ys + row * 128 + 64 * half;
        float* dst = Yg + ((size_t)(b * TT + q0 + row)) * CCH + h * DD + 64 * half;
        #pragma unroll
        for (int j = 0; j < 16; j++)
            *(float4*)(dst + 4 * j) = *(const float4*)(src + 4 * j);
    }
}

// ---------------------------------------------------------------------------
// Kernel 3: out = y @ W_proj + b_proj
// ---------------------------------------------------------------------------
__launch_bounds__(256)
__global__ void proj_kernel(const float* __restrict__ Y,
                            const float* __restrict__ W,
                            const float* __restrict__ bias,
                            float* __restrict__ out)
{
    __shared__ __align__(16) unsigned short As[2 * 128 * ASTR];
    __shared__ __align__(16) unsigned short Bs[2 * 128 * ASTR];
    const int nb = blockIdx.x, mb = blockIdx.y;

    v8f acc[2][4];
    gemm_core(Y, W, CCH, mb * 128, nb * 128, As, Bs, acc);

    const int tid = threadIdx.x, lane = tid & 31, wave = tid >> 5;
    const int l15 = lane & 15, hi = lane >> 4;
    const int wr = wave & 3, c0 = (wave >> 2) * 32;

    #pragma unroll
    for (int ci = 0; ci < 4; ci++) {
        const int ct  = c0 + ((ci & 1) * 16) + ((ci >> 1) * 64);
        const int col = nb * 128 + ct + l15;
        const float bv = bias[col];
        #pragma unroll
        for (int mi = 0; mi < 2; mi++)
            #pragma unroll
            for (int e = 0; e < 8; e++) {
                const int r = mb * 128 + 32 * wr + 16 * mi + e + 8 * hi;
                out[(size_t)r * CCH + col] = acc[mi][ci][e] + bv;
            }
    }
}

// ---------------------------------------------------------------------------
extern "C" void kernel_launch(void* const* d_in, const int* in_sizes, int n_in,
                              void* d_out, int out_size, void* d_ws, size_t ws_size,
                              hipStream_t stream)
{
    (void)in_sizes; (void)n_in; (void)out_size; (void)ws_size;
    const float* x   = (const float*)d_in[0];
    const float* Wat = (const float*)d_in[1];
    const float* bat = (const float*)d_in[2];
    const float* Wp  = (const float*)d_in[3];
    const float* bp  = (const float*)d_in[4];

    float* out  = (float*)d_out;                 // [B,T,C]
    float* kout = out + (size_t)BTC;             // [B,H,T,D] (roped k)
    float* vout = out + (size_t)2 * BTC;         // [B,H,T,D]
    float* qws  = (float*)d_ws;                  // [B,H,T,D] roped q (64 MB)
    float* yws  = qws + (size_t)BTC;             // [B,T,C] attention output (64 MB)

    qkv_rope_kernel<<<dim3(48, 64), 256, 0, stream>>>(x, Wat, bat, qws, kout, vout);
    attn_kernel<<<dim3(16, 16, 4), 256, 0, stream>>>(qws, kout, vout, yws);
    proj_kernel<<<dim3(16, 64), 256, 0, stream>>>(yws, Wp, bp, out);
}
